// Agent_6098853560698
// MI455X (gfx1250) — compile-verified, matrix operand order checked by reading, not run on hardware
//
#include <hip/hip_runtime.h>
#include <hip/hip_bf16.h>

// ---------------------------------------------------------------------------
// Types for CDNA5 WMMA (wave32): D(f32 16x16) = A(bf16 16x32) x B(bf16 32x16) + C
// ---------------------------------------------------------------------------
typedef __attribute__((ext_vector_type(16))) __bf16 v16bf;
typedef __attribute__((ext_vector_type(8)))  float  v8f;

#if __has_builtin(__builtin_amdgcn_s_wait_asynccnt)
#define HAVE_ASYNC_LDS 1
#else
#define HAVE_ASYNC_LDS 0
#endif

__device__ __forceinline__ __bf16 f2bf(float f) {
    // round-to-nearest-even fp32 -> bf16
    unsigned int u = __builtin_bit_cast(unsigned int, f);
    unsigned int r = (u + 0x7FFFu + ((u >> 16) & 1u)) >> 16;
    return __builtin_bit_cast(__bf16, (unsigned short)r);
}

#if HAVE_ASYNC_LDS
// CDNA5 async copy global -> LDS (tracked by ASYNCcnt).  vdst = per-lane LDS
// byte offset (low 32 bits of the generic LDS pointer == LDS offset per the
// aperture layout), vaddr = 64-bit global address in a VGPR pair.
__device__ __forceinline__ void async_load_b128(const __bf16* gsrc, __bf16* ldst) {
    unsigned int       loff = (unsigned int)(unsigned long long)ldst;
    unsigned long long gadr = (unsigned long long)gsrc;
    asm volatile("global_load_async_to_lds_b128 %0, %1, off"
                 :
                 : "v"(loff), "v"(gadr)
                 : "memory");
}
#endif

// Load a 16x32 bf16 fragment from LDS stored row-major with row stride 32.
// ISA 16-bit A layout: lane L -> row = L&15 ; K-octets (L>>4)*8 + {0..7, 16..23}
__device__ __forceinline__ v16bf frag_load(const __bf16* s, int rowbase, int lane) {
    int r  = rowbase + (lane & 15);
    int kb = (lane >> 4) * 8;
    const __bf16* p = s + r * 32 + kb;
    v16bf f;
#pragma unroll
    for (int j = 0; j < 8; ++j) { f[j] = p[j]; f[j + 8] = p[j + 16]; }
    return f;
}

// ---------------------------------------------------------------------------
// Implicit-GEMM convolution (NCHW in, OIHW weights pre-converted to bf16).
// GEMM view: M = OC, N = B*OH*OW, K = IC*KH*KW.
// Block: 128 threads = 4 waves. Block tile 64(M) x 64(N); wave w owns rows
// w*16..w*16+15 and all four 16-wide N sub-tiles (4 WMMAs per K-step of 32).
// Weight tile goes to LDS via async b128 loads when fully in-bounds.
// ---------------------------------------------------------------------------
__global__ void conv_wmma(const float* __restrict__ x, const __bf16* __restrict__ w,
                          float* __restrict__ out,
                          int B, int IC, int IH, int IW,
                          int OC, int OH, int OW,
                          int KH, int KW, int S, int P) {
    const int tid  = threadIdx.x;
    const int lane = tid & 31;
    const int wv   = tid >> 5;
    const int K    = IC * KH * KW;
    const int Ntot = B * OH * OW;
    const int n0   = blockIdx.x * 64;
    const int oc0  = blockIdx.y * 64;

    __shared__ __bf16 sA[64 * 32];  // weight tile [m][k]
    __shared__ __bf16 sB[64 * 32];  // im2col tile [n][k]

    // Cooperative-load coordinates: threads tid and tid+64 share row bn with
    // different 16-wide K halves, so each (row, k) covered exactly once.
    const int bn  = tid & 63;
    const int kh0 = (tid >> 6) * 16;
    const int gn  = n0 + bn;
    int bb = 0, oh = 0, ow = 0;
    const bool nvalid = gn < Ntot;
    if (nvalid) { bb = gn / (OH * OW); int r = gn % (OH * OW); oh = r / OW; ow = r - oh * OW; }

    const bool alignedK = (K & 31) == 0;  // stem (K=147) takes the guarded path
    v8f acc0 = {}, acc1 = {}, acc2 = {}, acc3 = {};

    for (int k0 = 0; k0 < K; k0 += 32) {
        const __bf16* wrow = w + (size_t)(oc0 + bn) * K + k0 + kh0;
#if HAVE_ASYNC_LDS
        if (alignedK) {
            async_load_b128(wrow,     &sA[bn * 32 + kh0]);
            async_load_b128(wrow + 8, &sA[bn * 32 + kh0 + 8]);
        } else
#endif
        {
#pragma unroll
            for (int j = 0; j < 16; ++j) {
                int k = k0 + kh0 + j;
                sA[bn * 32 + kh0 + j] = (k < K) ? wrow[j] : f2bf(0.0f);
            }
        }
        __builtin_prefetch(wrow + 64, 0, 1);  // weights two K tiles ahead (k0+64) -> L2

        // B: on-the-fly im2col, bf16 into LDS (needs conversion, stays VALU)
#pragma unroll
        for (int j = 0; j < 16; ++j) {
            int kl = kh0 + j;
            int k  = k0 + kl;
            float v = 0.0f;
            if (nvalid && k < K) {
                int ic = k / (KH * KW);
                int rr = k - ic * (KH * KW);
                int kh = rr / KW, kw = rr - kh * KW;
                int ih = oh * S - P + kh;
                int iw = ow * S - P + kw;
                if (ih >= 0 && ih < IH && iw >= 0 && iw < IW)
                    v = x[(((size_t)bb * IC + ic) * IH + ih) * IW + iw];
            }
            sB[bn * 32 + kl] = f2bf(v);
        }
#if HAVE_ASYNC_LDS
        if (alignedK) __builtin_amdgcn_s_wait_asynccnt(0);
#endif
        __syncthreads();
        v16bf af = frag_load(sA, wv * 16, lane);
        v16bf b0 = frag_load(sB, 0,  lane);
        v16bf b1 = frag_load(sB, 16, lane);
        v16bf b2 = frag_load(sB, 32, lane);
        v16bf b3 = frag_load(sB, 48, lane);
        acc0 = __builtin_amdgcn_wmma_f32_16x16x32_bf16(false, af, false, b0, (short)0, acc0, false, false);
        acc1 = __builtin_amdgcn_wmma_f32_16x16x32_bf16(false, af, false, b1, (short)0, acc1, false, false);
        acc2 = __builtin_amdgcn_wmma_f32_16x16x32_bf16(false, af, false, b2, (short)0, acc2, false, false);
        acc3 = __builtin_amdgcn_wmma_f32_16x16x32_bf16(false, af, false, b3, (short)0, acc3, false, false);
        __syncthreads();
    }

    // C layout: VGPR r -> lanes0-15: M=r, N=lane ; lanes16-31: M=8+r, N=lane-16
    int nIn   = lane & 15;
    int mBase = (lane >> 4) * 8;
#pragma unroll
    for (int sub = 0; sub < 4; ++sub) {
        v8f a = (sub == 0) ? acc0 : (sub == 1) ? acc1 : (sub == 2) ? acc2 : acc3;
        int g2 = n0 + sub * 16 + nIn;
        if (g2 >= Ntot) continue;
        int b2 = g2 / (OH * OW); int r2 = g2 % (OH * OW);
        int oh2 = r2 / OW, ow2 = r2 - oh2 * OW;
#pragma unroll
        for (int r = 0; r < 8; ++r) {
            int m = oc0 + wv * 16 + mBase + r;
            out[(((size_t)b2 * OC + m) * OH + oh2) * OW + ow2] = a[r];
        }
    }
}

// ---------------------------------------------------------------------------
// Generic GEMM: out[M][N] = A[M][K](f32) * W[N][K](bf16)^T (+bias[N]) (+extra[M][N]) (relu?)
// Block 128 threads = 4 waves; block tile 16(M) x 64(N); K stepped by 32.
// W tile uses async b128 loads when N%64==0 && K%32==0 (all LSTM/FC shapes).
// ---------------------------------------------------------------------------
__global__ void gemm_wmma(const float* __restrict__ A, const __bf16* __restrict__ W,
                          const float* __restrict__ bias, const float* __restrict__ extra,
                          float* __restrict__ out, int M, int N, int K, int relu) {
    const int tid  = threadIdx.x;
    const int lane = tid & 31;
    const int wv   = tid >> 5;
    const int m0   = blockIdx.y * 16;
    const int n0   = blockIdx.x * 64;

    __shared__ __bf16 sA[16 * 32];
    __shared__ __bf16 sW[64 * 32];

    const bool asyncOK = ((N & 63) == 0) && ((K & 31) == 0);
    v8f acc = {};

    for (int k0 = 0; k0 < K; k0 += 32) {
        {   // A: 16x32, thread -> m = tid&15, 4 k's at eighth (tid>>4)
            int m  = tid & 15;
            int kq = (tid >> 4) * 4;
            int gm = m0 + m;
#pragma unroll
            for (int j = 0; j < 4; ++j) {
                int k = k0 + kq + j;
                float v = (gm < M && k < K) ? A[(size_t)gm * K + k] : 0.0f;
                sA[m * 32 + kq + j] = f2bf(v);
            }
        }
        {   // W: 64x32, thread -> n = tid&63, 16 k's at half (tid>>6)
            int n   = tid & 63;
            int kh  = (tid >> 6) * 16;
            int gnn = n0 + n;
            const __bf16* wrow = W + (size_t)gnn * K + k0 + kh;
#if HAVE_ASYNC_LDS
            if (asyncOK) {
                async_load_b128(wrow,     &sW[n * 32 + kh]);
                async_load_b128(wrow + 8, &sW[n * 32 + kh + 8]);
            } else
#endif
            {
#pragma unroll
                for (int j = 0; j < 16; ++j) {
                    int k = k0 + kh + j;
                    sW[n * 32 + kh + j] = (gnn < N && k < K) ? wrow[j] : f2bf(0.0f);
                }
            }
            __builtin_prefetch(wrow + 64, 0, 1);
        }
#if HAVE_ASYNC_LDS
        if (asyncOK) __builtin_amdgcn_s_wait_asynccnt(0);
#endif
        __syncthreads();
        v16bf af = frag_load(sA, 0, lane);
        v16bf bf = frag_load(sW, wv * 16, lane);
        acc = __builtin_amdgcn_wmma_f32_16x16x32_bf16(false, af, false, bf, (short)0, acc, false, false);
        __syncthreads();
    }

    int nIn   = lane & 15;
    int mBase = (lane >> 4) * 8;
    int gn    = n0 + wv * 16 + nIn;
    if (gn >= N) return;
#pragma unroll
    for (int r = 0; r < 8; ++r) {
        int gm = m0 + mBase + r;
        if (gm >= M) continue;
        float v = acc[r];
        if (bias)  v += bias[gn];
        if (extra) v += extra[(size_t)gm * N + gn];
        if (relu)  v = fmaxf(v, 0.0f);
        out[(size_t)gm * N + gn] = v;
    }
}

// ---------------------------------------------------------------------------
// BatchNorm (training-mode batch stats) folded into per-channel scale/shift.
// ---------------------------------------------------------------------------
__global__ void bn_stats(const float* __restrict__ x, const float* __restrict__ g,
                         const float* __restrict__ b, float* __restrict__ scale,
                         float* __restrict__ shift, int B, int C, int HW) {
    int c = blockIdx.x;
    float s = 0.f, s2 = 0.f;
    int tot = B * HW;
    for (int i = threadIdx.x; i < tot; i += blockDim.x) {
        int bb = i / HW; int r = i - bb * HW;
        float v = x[((size_t)bb * C + c) * HW + r];
        s += v; s2 += v * v;
    }
    __shared__ float rs[256], rq[256];
    rs[threadIdx.x] = s; rq[threadIdx.x] = s2;
    __syncthreads();
    for (int o = 128; o > 0; o >>= 1) {
        if ((int)threadIdx.x < o) { rs[threadIdx.x] += rs[threadIdx.x + o]; rq[threadIdx.x] += rq[threadIdx.x + o]; }
        __syncthreads();
    }
    if (threadIdx.x == 0) {
        float cnt  = (float)tot;
        float mean = rs[0] / cnt;
        float var  = rq[0] / cnt - mean * mean;
        float sc   = g[c] * rsqrtf(var + 1e-5f);
        scale[c] = sc;
        shift[c] = b[c] - mean * sc;
    }
}

__global__ void bn_apply(const float* __restrict__ x, const float* __restrict__ scale,
                         const float* __restrict__ shift, const float* __restrict__ skip,
                         float* __restrict__ out, int C, int HW, long total, int relu) {
    long i = blockIdx.x * (long)blockDim.x + threadIdx.x;
    if (i >= total) return;
    int c = (int)((i / HW) % C);
    float v = scale[c] * x[i] + shift[c];
    if (skip) v += skip[i];
    if (relu) v = fmaxf(v, 0.0f);
    out[i] = v;
}

__global__ void maxpool_k(const float* __restrict__ x, float* __restrict__ out,
                          int B, int C, int IH, int IW, int OH, int OW) {
    long i = blockIdx.x * (long)blockDim.x + threadIdx.x;
    long total = (long)B * C * OH * OW;
    if (i >= total) return;
    int ow = (int)(i % OW); long r = i / OW;
    int oh = (int)(r % OH); r /= OH;
    int c  = (int)(r % C);
    int b  = (int)(r / C);
    float m = -3.4e38f;
    for (int kh = 0; kh < 3; ++kh) {
        int ih = oh * 2 - 1 + kh;
        if (ih < 0 || ih >= IH) continue;
        for (int kw = 0; kw < 3; ++kw) {
            int iw = ow * 2 - 1 + kw;
            if (iw < 0 || iw >= IW) continue;
            m = fmaxf(m, x[(((size_t)b * C + c) * IH + ih) * IW + iw]);
        }
    }
    out[i] = m;
}

// ---------------------------------------------------------------------------
// LSTM cell: g holds (i,f,g,o) pre-activations (16 x 4096); PyTorch gate order.
// ---------------------------------------------------------------------------
__device__ __forceinline__ float sigm(float x) { return 1.0f / (1.0f + __expf(-x)); }

__global__ void lstm_cell(const float* __restrict__ g, float* __restrict__ h,
                          float* __restrict__ c, float* __restrict__ hs, int t) {
    int i = blockIdx.x * blockDim.x + threadIdx.x;
    if (i >= 16 * 1024) return;
    int b = i >> 10, n = i & 1023;
    const float* gr = g + (size_t)b * 4096;
    float ig = sigm(gr[n]);
    float fg = sigm(gr[n + 1024]);
    float gg = tanhf(gr[n + 2048]);
    float og = sigm(gr[n + 3072]);
    float cv = fg * c[i] + ig * gg;
    float hv = og * tanhf(cv);
    c[i] = cv; h[i] = hv;
    hs[((size_t)b * 196 + t) * 1024 + n] = hv;  // (B,196,H) layout
}

__global__ void values_head(const float* __restrict__ hs, const float* __restrict__ w2,
                            const float* __restrict__ b2, float* __restrict__ out) {
    int row = blockIdx.x;  // 0..3135
    float s = 0.f;
    for (int k = threadIdx.x; k < 1024; k += 256) s += hs[(size_t)row * 1024 + k] * w2[k];
    __shared__ float red[256];
    red[threadIdx.x] = s;
    __syncthreads();
    for (int o = 128; o > 0; o >>= 1) {
        if ((int)threadIdx.x < o) red[threadIdx.x] += red[threadIdx.x + o];
        __syncthreads();
    }
    if (threadIdx.x == 0) out[row] = red[0] + b2[0];
}

__global__ void to_bf16_k(const float* __restrict__ x, __bf16* __restrict__ y, long n) {
    long i = blockIdx.x * (long)blockDim.x + threadIdx.x;
    if (i < n) y[i] = f2bf(x[i]);
}
__global__ void add_vec(const float* a, const float* b, float* o, int n) {
    int i = blockIdx.x * blockDim.x + threadIdx.x;
    if (i < n) o[i] = a[i] + b[i];
}
__global__ void zero_k(float* p, long n) {
    long i = blockIdx.x * (long)blockDim.x + threadIdx.x;
    if (i < n) p[i] = 0.0f;
}

// ---------------------------------------------------------------------------
// Host orchestration
// ---------------------------------------------------------------------------
extern "C" void kernel_launch(void* const* d_in, const int* in_sizes, int n_in,
                              void* d_out, int out_size, void* d_ws, size_t ws_size,
                              hipStream_t stream) {
    (void)in_sizes; (void)n_in; (void)out_size; (void)ws_size;
    const int B = 16;
    const float* x = (const float*)d_in[0];

    // --- walk jax pytree leaf order: 'layers' (sorted keys per block) then 'stem' ---
    struct BlockP { const float *w1,*g1,*b1,*w2,*g2,*b2,*wd,*gd,*bd; bool ds; int ic, oc, s; };
    BlockP blk[16];
    const int cfg_oc[4] = {64, 128, 256, 512};
    const int cfg_nb[4] = {3, 4, 6, 3};
    int idx = 1;
    {
        int bi = 0, in_c = 64;
        for (int L = 0; L < 4; ++L)
            for (int k = 0; k < cfg_nb[L]; ++k) {
                BlockP p{};
                int s = (k == 0) ? (L == 0 ? 1 : 2) : 1;
                p.ds = (s != 1) || (in_c != cfg_oc[L]);
                p.ic = in_c; p.oc = cfg_oc[L]; p.s = s;
                p.b1 = (const float*)d_in[idx++];
                p.b2 = (const float*)d_in[idx++];
                if (p.ds) p.bd = (const float*)d_in[idx++];
                p.g1 = (const float*)d_in[idx++];
                p.g2 = (const float*)d_in[idx++];
                if (p.ds) p.gd = (const float*)d_in[idx++];
                p.w1 = (const float*)d_in[idx++];
                p.w2 = (const float*)d_in[idx++];
                if (p.ds) p.wd = (const float*)d_in[idx++];
                blk[bi++] = p;
                in_c = cfg_oc[L];
            }
    }
    const float* stem_b = (const float*)d_in[idx++];
    const float* stem_g = (const float*)d_in[idx++];
    const float* stem_w = (const float*)d_in[idx++];
    const float* w1fc = (const float*)d_in[idx++];
    const float* b1fc = (const float*)d_in[idx++];
    const float* w_ih = (const float*)d_in[idx++];
    const float* w_hh = (const float*)d_in[idx++];
    const float* b_ih = (const float*)d_in[idx++];
    const float* b_hh = (const float*)d_in[idx++];
    const float* wh1  = (const float*)d_in[idx++];
    const float* bh1  = (const float*)d_in[idx++];
    const float* wh2  = (const float*)d_in[idx++];
    const float* bh2  = (const float*)d_in[idx++];

    // --- workspace bump allocator ---
    size_t off = 0;
    auto alloc = [&](size_t elems, size_t esz) -> void* {
        void* p = (char*)d_ws + off;
        off = (off + elems * esz + 255) & ~(size_t)255;
        return p;
    };
    auto cvt = [&](const float* src, long n) -> __bf16* {
        __bf16* dst = (__bf16*)alloc((size_t)n, 2);
        to_bf16_k<<<dim3((unsigned)((n + 255) / 256)), 256, 0, stream>>>(src, dst, n);
        return dst;
    };

    // bf16 weight copies
    __bf16* stem_wb = cvt(stem_w, 64L * 3 * 49);
    __bf16 *w1b[16], *w2b[16], *wdb[16];
    for (int i = 0; i < 16; ++i) {
        w1b[i] = cvt(blk[i].w1, (long)blk[i].oc * blk[i].ic * 9);
        w2b[i] = cvt(blk[i].w2, (long)blk[i].oc * blk[i].oc * 9);
        wdb[i] = blk[i].ds ? cvt(blk[i].wd, (long)blk[i].oc * blk[i].ic) : nullptr;
    }
    __bf16* w1fb = cvt(w1fc, 1024L * 25088);
    __bf16* wihb = cvt(w_ih, 4096L * 1024);
    __bf16* whhb = cvt(w_hh, 4096L * 1024);
    __bf16* wh1b = cvt(wh1, 197L * 1024);

    // activation buffers
    float* bufStem = (float*)alloc(16ull * 64 * 112 * 112, 4);
    float* X  = (float*)alloc(16ull * 64 * 56 * 56, 4);
    float* T1 = (float*)alloc(16ull * 64 * 56 * 56, 4);
    float* T2 = (float*)alloc(16ull * 64 * 56 * 56, 4);
    float* TD = (float*)alloc(16ull * 64 * 56 * 56, 4);
    float* sc1 = (float*)alloc(512, 4); float* sh1 = (float*)alloc(512, 4);
    float* scd = (float*)alloc(512, 4); float* shd = (float*)alloc(512, 4);
    float* xin  = (float*)alloc(16 * 1024, 4);
    float* bihh = (float*)alloc(4096, 4);
    float* gx   = (float*)alloc(16 * 4096, 4);
    float* gbuf = (float*)alloc(16 * 4096, 4);
    float* hbuf = (float*)alloc(16 * 1024, 4);
    float* cbuf = (float*)alloc(16 * 1024, 4);
    float* hs   = (float*)alloc(3136ull * 1024, 4);

    auto nblk = [](long n, int bs) { return (unsigned)((n + bs - 1) / bs); };

    // --- stem: conv7x7 s2 p3 -> BN -> ReLU -> maxpool3x3 s2 p1 ---
    {
        int Nt = B * 112 * 112;
        conv_wmma<<<dim3(nblk(Nt, 64), 1), 128, 0, stream>>>(x, stem_wb, bufStem,
            B, 3, 224, 224, 64, 112, 112, 7, 7, 2, 3);
        bn_stats<<<64, 256, 0, stream>>>(bufStem, stem_g, stem_b, sc1, sh1, B, 64, 112 * 112);
        long tot = (long)Nt * 64;
        bn_apply<<<nblk(tot, 256), 256, 0, stream>>>(bufStem, sc1, sh1, nullptr, bufStem, 64, 112 * 112, tot, 1);
        long ptot = 16L * 64 * 56 * 56;
        maxpool_k<<<nblk(ptot, 256), 256, 0, stream>>>(bufStem, X, B, 64, 112, 112, 56, 56);
    }

    // --- residual blocks ---
    int H = 56, Wd = 56;
    for (int bi = 0; bi < 16; ++bi) {
        const BlockP& p = blk[bi];
        int s = p.s, oc = p.oc, ic = p.ic;
        int OHb = (s == 2) ? H / 2 : H;
        int OWb = (s == 2) ? Wd / 2 : Wd;
        int Nt = B * OHb * OWb;
        long tot = (long)Nt * oc;

        conv_wmma<<<dim3(nblk(Nt, 64), oc / 64), 128, 0, stream>>>(X, w1b[bi], T1,
            B, ic, H, Wd, oc, OHb, OWb, 3, 3, s, 1);
        bn_stats<<<oc, 256, 0, stream>>>(T1, p.g1, p.b1, sc1, sh1, B, oc, OHb * OWb);
        bn_apply<<<nblk(tot, 256), 256, 0, stream>>>(T1, sc1, sh1, nullptr, T1, oc, OHb * OWb, tot, 1);

        conv_wmma<<<dim3(nblk(Nt, 64), oc / 64), 128, 0, stream>>>(T1, w2b[bi], T2,
            B, oc, OHb, OWb, oc, OHb, OWb, 3, 3, 1, 1);
        bn_stats<<<oc, 256, 0, stream>>>(T2, p.g2, p.b2, sc1, sh1, B, oc, OHb * OWb);

        const float* skip = X;
        if (p.ds) {
            conv_wmma<<<dim3(nblk(Nt, 64), oc / 64), 128, 0, stream>>>(X, wdb[bi], TD,
                B, ic, H, Wd, oc, OHb, OWb, 1, 1, s, 0);
            bn_stats<<<oc, 256, 0, stream>>>(TD, p.gd, p.bd, scd, shd, B, oc, OHb * OWb);
            bn_apply<<<nblk(tot, 256), 256, 0, stream>>>(TD, scd, shd, nullptr, TD, oc, OHb * OWb, tot, 0);
            skip = TD;
        }
        // out = relu(bn(conv2) + skip) -> T1 (free after conv2), then swap into X
        bn_apply<<<nblk(tot, 256), 256, 0, stream>>>(T2, sc1, sh1, skip, T1, oc, OHb * OWb, tot, 1);
        float* tmp = X; X = T1; T1 = tmp;
        H = OHb; Wd = OWb;
    }
    // X now holds (16, 512, 7, 7) == feat rows (16, 25088)

    // --- FC1: xin = relu(feat @ w1^T + b1) ---
    gemm_wmma<<<dim3(nblk(1024, 64), 1), 128, 0, stream>>>(X, w1fb, b1fc, nullptr, xin, 16, 1024, 25088, 1);

    // --- gx = xin @ w_ih^T + (b_ih + b_hh) ---
    add_vec<<<16, 256, 0, stream>>>(b_ih, b_hh, bihh, 4096);
    gemm_wmma<<<dim3(nblk(4096, 64), 1), 128, 0, stream>>>(xin, wihb, bihh, nullptr, gx, 16, 4096, 1024, 0);

    // --- LSTM: 196 sequential steps (bf16 w_hh resident in L2) ---
    zero_k<<<64, 256, 0, stream>>>(hbuf, 16 * 1024);
    zero_k<<<64, 256, 0, stream>>>(cbuf, 16 * 1024);
    for (int t = 0; t < 196; ++t) {
        gemm_wmma<<<dim3(nblk(4096, 64), 1), 128, 0, stream>>>(hbuf, whhb, nullptr, gx, gbuf, 16, 4096, 1024, 0);
        lstm_cell<<<64, 256, 0, stream>>>(gbuf, hbuf, cbuf, hs, t);
    }

    // --- heads ---
    float* out_table = (float*)d_out;
    float* out_vals  = out_table + 3136L * 197;
    gemm_wmma<<<dim3(nblk(197, 64), 3136 / 16), 128, 0, stream>>>(hs, wh1b, bh1, nullptr, out_table, 3136, 197, 1024, 0);
    values_head<<<3136, 256, 0, stream>>>(hs, wh2, bh2, out_vals);
}
